// MoeLongformerScorer_82317343195591
// MI455X (gfx1250) — matrix operand order repaired and checked by reading, not compiled, not run on hardware
//
#include <hip/hip_runtime.h>
#include <hip/hip_bf16.h>

#define B_ 2
#define S_ 2048
#define H_ 768
#define I_ 3072
#define E_ 7
#define T_ (B_*S_)       // 4096 tokens
#define MT 32            // tokens per workgroup (2 x 16-row subtiles)
#define NT1 (I_/16)      // 192  n-tiles in I  (GEMM1 B)
#define NT2 (H_/16)      // 48   n-tiles in H  (GEMM2 B)
#define KS1 (H_/32)      // 24   k-steps for GEMM1
#define KS2 (I_/32)      // 96   k-steps for GEMM2 (global)

typedef __attribute__((ext_vector_type(16))) _Float16 v16h;
typedef __attribute__((ext_vector_type(8)))  float    v8f;

// 16-bit A-matrix (16x32) wave32 fragment layout (ISA 7.12.2):
// lane L: M = L%16 ; kbase = (L<16)?0:8
//   frag[2d+j] = K = kbase+2d+j (d<4) or 16+kbase+2(d-4)+j (d>=4)
__device__ __forceinline__ int frag_K_A(int L, int d, int j) {
  int kbase = (L < 16) ? 0 : 8;
  return (d < 4) ? (kbase + 2*d + j) : (16 + kbase + 2*(d - 4) + j);
}
// 16-bit B-matrix (32x16) layout (per ISA B description):
// lane L: N = L%16 ; kbase = (L<16)?0:8 ; pos<8 -> K = kbase+pos, pos>=8 -> K = 16+kbase+pos-8
__device__ __forceinline__ int frag_K_B(int L, int pos) {
  int kbase = (L < 16) ? 0 : 8;
  return (pos < 8) ? (kbase + pos) : (16 + kbase + pos - 8);
}

// ---------------- zero output accumulator region ----------------
__global__ void zero_out_kernel(float* __restrict__ out, int n) {
  int i = blockIdx.x * blockDim.x + threadIdx.x;
  if (i < n) out[i] = 0.0f;
}

// ---------------- gating: exact fp32, one wave per token ----------------
__global__ void gate_kernel(const float* __restrict__ x, const float* __restrict__ wg,
                            const float* __restrict__ bg, float* __restrict__ gate) {
  int wave = threadIdx.x >> 5;
  int lane = threadIdx.x & 31;
  int t = blockIdx.x * (blockDim.x >> 5) + wave;
  if (t >= T_) return;
  const float* xr = x + (size_t)t * H_;
  float lg[E_];
  for (int e = 0; e < E_; ++e) {
    float acc = 0.0f;
    for (int k = lane; k < H_; k += 32) acc += xr[k] * wg[k * E_ + e];
    for (int off = 16; off > 0; off >>= 1) acc += __shfl_down(acc, off);
    lg[e] = acc + bg[e];   // valid in lane 0 after reduction
  }
  if (lane == 0) {
    int bi = 0; float best = lg[0];
    for (int e = 1; e < E_; ++e) if (lg[e] > best) { best = lg[e]; bi = e; }
    int si = (bi == 0) ? 1 : 0; float second = lg[si];
    for (int e = 0; e < E_; ++e)
      if (e != bi && lg[e] > second) { second = lg[e]; si = e; }
    float ex = __expf(second - best);
    float inv = 1.0f / (1.0f + ex);
    float* gr = gate + (size_t)t * E_;
    for (int e = 0; e < E_; ++e) gr[e] = 0.0f;
    gr[bi] = inv;
    gr[si] = ex * inv;
  }
}

// ---------------- one-time fp32 -> f16 weight swizzle ----------------
// w1s block layout: [(e*NT1 + nt)*KS1 + ks][lane(32)][pos(16)] , 1KB blocks
__global__ void convert_w1_kernel(const float* __restrict__ w1, _Float16* __restrict__ w1s) {
  long idx = (long)blockIdx.x * blockDim.x + threadIdx.x;
  const long total = (long)E_ * H_ * I_;
  if (idx >= total) return;
  int pos = idx & 15;
  int L   = (int)((idx >> 4) & 31);
  long blk = idx >> 9;
  int ks = (int)(blk % KS1);
  long tmp = blk / KS1;
  int nt = (int)(tmp % NT1);
  int e  = (int)(tmp / NT1);
  int k = ks * 32 + frag_K_B(L, pos);      // row in H
  int n = nt * 16 + (L & 15);              // col in I
  w1s[idx] = (_Float16)w1[((long)e * H_ + k) * I_ + n];
}

// w2s block layout: [(e*NT2 + nt)*KS2 + ks][lane(32)][pos(16)]
__global__ void convert_w2_kernel(const float* __restrict__ w2, _Float16* __restrict__ w2s) {
  long idx = (long)blockIdx.x * blockDim.x + threadIdx.x;
  const long total = (long)E_ * I_ * H_;
  if (idx >= total) return;
  int pos = idx & 15;
  int L   = (int)((idx >> 4) & 31);
  long blk = idx >> 9;
  int ks = (int)(blk % KS2);
  long tmp = blk / KS2;
  int nt = (int)(tmp % NT2);
  int e  = (int)(tmp / NT2);
  int k = ks * 32 + frag_K_B(L, pos);      // row in I
  int n = nt * 16 + (L & 15);              // col in H
  w2s[idx] = (_Float16)w2[((long)e * I_ + k) * H_ + n];
}

// ---------------- fused dual-GEMM MoE FFN ----------------
// grid (T_/MT, E_), 512 threads = 16 waves.
// GEMM1: wave w -> msub = w>>3 (16-token subtile), c8 = w&7 (16-col group of chunk)
// GEMM2: wave w -> n-tiles {3w, 3w+1, 3w+2} x both msubs (each w2 block loaded once/WG)
__global__ __launch_bounds__(512)
void moe_ffn_kernel(const float* __restrict__ x,
                    const float* __restrict__ b1,
                    const float* __restrict__ b2,
                    const _Float16* __restrict__ w1s,
                    const _Float16* __restrict__ w2s,
                    const float* __restrict__ gate,
                    float* __restrict__ out) {
  __shared__ __align__(32) _Float16 xs[2 * KS1 * 512];      // 48 KB, A-frag order
  __shared__ __align__(32) _Float16 hs[2 * 2 * 4 * 512];    // 16 KB, [buf][msub][ks][...]

  const int e    = blockIdx.y;
  const int tok0 = blockIdx.x * MT;
  const int tid  = threadIdx.x;
  const int wave = tid >> 5, lane = tid & 31;
  const int msub = wave >> 3;   // 0..1   (GEMM1 row subtile)
  const int c8   = wave & 7;    // 0..7   (GEMM1 column group)

  // stage x tile fp32 -> f16 directly into A-fragment layout
  for (int idx = tid; idx < 2 * KS1 * 512; idx += 512) {
    int pos = idx & 15, L = (idx >> 4) & 31, blk = idx >> 9;
    int ks = blk % KS1, ms = blk / KS1;
    int k = ks * 32 + frag_K_A(L, pos >> 1, pos & 1);
    int m = ms * 16 + (L & 15);
    xs[idx] = (_Float16)x[(size_t)(tok0 + m) * H_ + k];
  }
  __syncthreads();

  v8f acc[6];   // [msub*3 + t]
  {
    v8f z = {};
    #pragma unroll
    for (int t = 0; t < 6; ++t) acc[t] = z;
  }

  // precompute hs scatter pattern (fixed per lane)
  const int nloc = c8 * 16 + (lane & 15);          // column within 128-chunk
  const int K2 = nloc & 31, hks = nloc >> 5;
  const int kb2 = ((K2 & 15) < 8) ? 0 : 8;
  const int Lgrp = (kb2 == 0) ? 0 : 16;
  const int rr = K2 - kb2;
  const int d2 = (rr < 8) ? (rr >> 1) : (4 + ((rr - 16) >> 1));
  const int pos2 = 2 * d2 + (K2 & 1);

  for (int io = 0; io < I_; io += 128) {
    const int buf = (io >> 7) & 1;
    // ---- GEMM1: h tile [msub, io + 16*c8 .. +16), K = H ----
    v8f hc = {};
    const _Float16* bptr =
        w1s + ((size_t)(e * NT1 + (io >> 4) + c8) * KS1) * 512 + lane * 16;
    // warm WGP-level caches with next chunk's w1 blocks (high locality -> WGP scope)
    __builtin_prefetch(bptr + (size_t)KS1 * 512 * 8, 0, 3);
    #pragma unroll
    for (int ks = 0; ks < KS1; ++ks) {
      v16h a = *(const v16h*)&xs[((msub * KS1 + ks) * 32 + lane) * 16];
      v16h b = *(const v16h*)(bptr + (size_t)ks * 512);
      hc = __builtin_amdgcn_wmma_f32_16x16x32_f16(false, a, false, b,
                                                  (short)0, hc, false, false);
    }
    // bias + exact GELU, scatter into hs[buf] (A-frag order for GEMM2)
    const float b1v = b1[e * I_ + io + nloc];
    #pragma unroll
    for (int r = 0; r < 8; ++r) {
      int mrow = r + ((lane < 16) ? 0 : 8);
      float hv = hc[r] + b1v;
      float g = 0.5f * hv * (1.0f + erff(hv * 0.70710678118654752f));
      hs[((((buf * 2 + msub) * 4 + hks) * 32) + (mrow + Lgrp)) * 16 + pos2] = (_Float16)g;
    }
    __syncthreads();   // single barrier per chunk (hs double-buffered)

    // ---- GEMM2: n-tiles {3w..3w+2} x both msubs, K = this 128-chunk ----
    #pragma unroll
    for (int ks = 0; ks < 4; ++ks) {
      v16h a0 = *(const v16h*)&hs[(((buf * 2 + 0) * 4 + ks) * 32 + lane) * 16];
      v16h a1 = *(const v16h*)&hs[(((buf * 2 + 1) * 4 + ks) * 32 + lane) * 16];
      int kglob = (io >> 5) + ks;
      #pragma unroll
      for (int t = 0; t < 3; ++t) {
        int nt2 = wave * 3 + t;
        const _Float16* b2ptr = w2s +
            ((size_t)(e * NT2 + nt2) * KS2 + kglob) * 512 + lane * 16;
        if (ks == 0) __builtin_prefetch(b2ptr + 4 * 512, 0, 3);  // next chunk's k-steps
        v16h b = *(const v16h*)b2ptr;
        acc[t]     = __builtin_amdgcn_wmma_f32_16x16x32_f16(false, a0, false, b,
                                                            (short)0, acc[t], false, false);
        acc[3 + t] = __builtin_amdgcn_wmma_f32_16x16x32_f16(false, a1, false, b,
                                                            (short)0, acc[3 + t], false, false);
      }
    }
  }

  // ---- epilogue: + b2, gate-weighted atomic combine (skip gate==0) ----
  #pragma unroll
  for (int ms = 0; ms < 2; ++ms) {
    #pragma unroll
    for (int t = 0; t < 3; ++t) {
      int hcol = (wave * 3 + t) * 16 + (lane & 15);
      float b2v = b2[e * H_ + hcol];
      #pragma unroll
      for (int r = 0; r < 8; ++r) {
        int m = ms * 16 + r + ((lane < 16) ? 0 : 8);
        int token = tok0 + m;
        float g = gate[(size_t)token * E_ + e];
        if (g != 0.0f)
          atomicAdd(&out[(size_t)token * H_ + hcol], g * (acc[ms * 3 + t][r] + b2v));
      }
    }
  }
}

extern "C" void kernel_launch(void* const* d_in, const int* in_sizes, int n_in,
                              void* d_out, int out_size, void* d_ws, size_t ws_size,
                              hipStream_t stream) {
  const float* x  = (const float*)d_in[0];
  const float* wg = (const float*)d_in[1];
  const float* bg = (const float*)d_in[2];
  const float* w1 = (const float*)d_in[3];
  const float* b1 = (const float*)d_in[4];
  const float* w2 = (const float*)d_in[5];
  const float* b2 = (const float*)d_in[6];

  float* out  = (float*)d_out;               // [B,S,H]
  float* gate = out + (size_t)T_ * H_;       // [B,S,E] (output #2)

  _Float16* w1s = (_Float16*)d_ws;                       // 33 MB
  _Float16* w2s = w1s + (size_t)E_ * H_ * I_;            // 33 MB

  {
    int n = T_ * H_;
    zero_out_kernel<<<(n + 255) / 256, 256, 0, stream>>>(out, n);
  }
  gate_kernel<<<T_ / 8, 256, 0, stream>>>(x, wg, bg, gate);

  {
    long total = (long)E_ * H_ * I_;
    int blocks = (int)((total + 255) / 256);
    convert_w1_kernel<<<blocks, 256, 0, stream>>>(w1, w1s);
    convert_w2_kernel<<<blocks, 256, 0, stream>>>(w2, w2s);
  }

  dim3 grid(T_ / MT, E_);
  moe_ffn_kernel<<<grid, 512, 0, stream>>>(x, b1, b2, w1s, w2s, gate, out);
}
